// DeformableConv2d1v_5231270167096
// MI455X (gfx1250) — matrix operand order repaired and checked by reading, not compiled, not run on hardware
//
#include <hip/hip_runtime.h>
#include <hip/hip_bf16.h>

// ---------------------------------------------------------------------------
// Deformable Conv v1 on MI455X (gfx1250, wave32, WMMA)
//   x:     (8, 64, 112, 112) f32
//   w_off: (18, 64, 3, 3)    f32   -> packed bf16 [32][576]  (k = tap*64 + c)
//   b_off: (18,)             f32
//   w:     (128, 64, 3, 3)   f32   -> packed bf16 [128][576] (k = tap*64 + c)
//   out:   (8, 128, 112, 112) f32
// ---------------------------------------------------------------------------

#define CIN   64
#define COUT  128
#define HW    112
#define KDIM  576          // 9 taps * 64 channels
#define ASTR  584          // padded LDS row stride (bf16 elems), 16B-aligned

typedef __attribute__((ext_vector_type(16))) __bf16 v16bf;
typedef __attribute__((ext_vector_type(8)))  __bf16 v8bf;
typedef __attribute__((ext_vector_type(8)))  float  v8f;
typedef __attribute__((ext_vector_type(4)))  float  v4f;

__device__ __forceinline__ int iclamp(int v, int lo, int hi) {
    return v < lo ? lo : (v > hi ? hi : v);
}

// --- pack main conv weights: wB[oc][tap*64 + c] = w[oc][c][tap] ------------
__global__ __launch_bounds__(256) void pack_w_kernel(
    const float* __restrict__ w, __bf16* __restrict__ wB)
{
    int gid = blockIdx.x * 256 + threadIdx.x;      // 128*576 = 73728 = 288*256
    int oc  = gid / KDIM;
    int j   = gid % KDIM;
    int tap = j >> 6;
    int c   = j & 63;
    wB[gid] = (__bf16)w[(oc * CIN + c) * 9 + tap];
}

// --- pack offset conv weights (pad 18 -> 32 out channels with zeros) -------
__global__ __launch_bounds__(256) void pack_woff_kernel(
    const float* __restrict__ w_off, __bf16* __restrict__ wOffB)
{
    int gid = blockIdx.x * 256 + threadIdx.x;      // 32*576 = 18432 = 72*256
    int oc  = gid / KDIM;
    int j   = gid % KDIM;
    int tap = j >> 6;
    int c   = j & 63;
    float v = (oc < 18) ? w_off[(oc * CIN + c) * 9 + tap] : 0.0f;
    wOffB[gid] = (__bf16)v;
}

// --- fused: offset-conv (WMMA) -> bilinear sample -> main GEMM (WMMA) ------
__global__ __launch_bounds__(256) void deform_wmma_kernel(
    const float* __restrict__ x,
    const float* __restrict__ b_off,
    const __bf16* __restrict__ wOffB,
    const __bf16* __restrict__ wB,
    float* __restrict__ out)
{
    __shared__ __bf16 As[16][ASTR];     // shared A tile (im2col, then sampled)
    __shared__ float  sOff[16][20];     // per-pixel offsets, 18 channels
    __shared__ int    sY[144];
    __shared__ int    sX[144];
    __shared__ float  sWy[144];
    __shared__ float  sWx[144];

    const int tid = threadIdx.x;
    const int blk = blockIdx.x;                 // 8 * 112 * 7 = 6272 blocks
    const int xt  = blk % 7;
    const int y   = (blk / 7) % HW;
    const int b   = blk / (7 * HW);
    const int x0  = xt * 16;

    const int lane = tid & 31;
    const int wv   = tid >> 5;                  // wave 0..7
    const int row  = lane & 15;
    const int half = lane >> 4;

    const int p    = tid & 15;                  // pixel within tile
    const int krow = tid >> 4;                  // 0..15

    // ---- Phase A1: build im2col A tile (zero-padded 3x3 neighborhood) ----
    {
        const float* xb = x + (size_t)b * CIN * HW * HW;
        #pragma unroll 4
        for (int it = 0; it < 36; ++it) {
            int k   = it * 16 + krow;           // 0..575
            int tap = k >> 6;
            int c   = k & 63;
            int ky  = tap / 3, kx = tap % 3;
            int yy  = y + ky - 1;
            int xx  = x0 + p + kx - 1;
            bool ok = ((unsigned)yy < HW) & ((unsigned)xx < HW);
            float v = ok ? xb[(size_t)c * HW * HW + yy * HW + xx] : 0.0f;
            As[p][k] = (__bf16)v;
        }
    }
    __syncthreads();

    // ---- Phase A2: offset conv via WMMA (waves 0,1 -> 32 padded oc) ------
    if (wv < 2) {
        v8f acc = {};
        const int oc = wv * 16 + row;           // column of B
        const __bf16* bbase = wOffB + (size_t)oc * KDIM + half * 16;
        for (int kt = 0; kt < 18; ++kt) {
            int k0 = kt * 32;
            v8bf alo = *(const v8bf*)&As[row][k0 + half * 16];
            v8bf ahi = *(const v8bf*)&As[row][k0 + half * 16 + 8];
            v16bf a;
            #pragma unroll
            for (int i = 0; i < 8; ++i) { a[i] = alo[i]; a[i + 8] = ahi[i]; }
            v16bf bf = *(const v16bf*)(bbase + k0);
            acc = __builtin_amdgcn_wmma_f32_16x16x32_bf16(
                      false, a, false, bf, (short)0, acc, false, false);
        }
        if (oc < 18) {
            float bo = b_off[oc];
            #pragma unroll
            for (int r = 0; r < 8; ++r)
                sOff[r + 8 * half][oc] = acc[r] + bo;   // D: M = r + 8*half
        }
    }
    __syncthreads();

    // ---- Phase B1: bilinear coordinates for 16 px * 9 taps ---------------
    if (tid < 144) {
        int pp  = tid / 9;
        int tap = tid % 9;
        int ky  = tap / 3, kx = tap % 3;
        float dy  = sOff[pp][2 * tap];
        float dx  = sOff[pp][2 * tap + 1];
        float py  = dy + (float)(y - 1 + ky);
        float pxf = dx + (float)(x0 + pp - 1 + kx);
        float fy = floorf(py);
        float fx = floorf(pxf);
        sY[tid]  = (int)fy;
        sX[tid]  = (int)fx;
        sWy[tid] = py - fy;
        sWx[tid] = pxf - fx;
    }
    __syncthreads();

    // ---- Phase B2: gather sampled A tile (overwrite As) ------------------
    {
        const float* xb = x + (size_t)b * CIN * HW * HW;
        #pragma unroll 4
        for (int it = 0; it < 36; ++it) {
            int k    = it * 16 + krow;
            int tap  = k >> 6;
            int c    = k & 63;
            int pair = p * 9 + tap;
            int yi = sY[pair];
            int xi = sX[pair];
            float wy = sWy[pair];
            float wx = sWx[pair];
            const float* xc = xb + (size_t)c * HW * HW;
            float m00 = (float)(((unsigned)yi     < HW) & ((unsigned)xi     < HW));
            float m01 = (float)(((unsigned)yi     < HW) & ((unsigned)(xi+1) < HW));
            float m10 = (float)(((unsigned)(yi+1) < HW) & ((unsigned)xi     < HW));
            float m11 = (float)(((unsigned)(yi+1) < HW) & ((unsigned)(xi+1) < HW));
            int y0 = iclamp(yi,     0, HW - 1);
            int y1 = iclamp(yi + 1, 0, HW - 1);
            int xc0 = iclamp(xi,     0, HW - 1);
            int xc1 = iclamp(xi + 1, 0, HW - 1);
            float v00 = xc[y0 * HW + xc0] * m00;
            float v01 = xc[y0 * HW + xc1] * m01;
            float v10 = xc[y1 * HW + xc0] * m10;
            float v11 = xc[y1 * HW + xc1] * m11;
            float omwy = 1.0f - wy, omwx = 1.0f - wx;
            float val = v00 * omwy * omwx + v01 * omwy * wx
                      + v10 * wy * omwx + v11 * wy * wx;
            As[p][k] = (__bf16)val;
        }
    }
    __syncthreads();

    // ---- Phase C: main GEMM, each wave owns a 16-oc N tile ---------------
    v8f acc = {};
    {
        const int col = wv * 16 + row;
        const __bf16* bbase = wB + (size_t)col * KDIM + half * 16;
        for (int kt = 0; kt < 18; ++kt) {
            int k0 = kt * 32;
            v8bf alo = *(const v8bf*)&As[row][k0 + half * 16];
            v8bf ahi = *(const v8bf*)&As[row][k0 + half * 16 + 8];
            v16bf a;
            #pragma unroll
            for (int i = 0; i < 8; ++i) { a[i] = alo[i]; a[i + 8] = ahi[i]; }
            v16bf bf = *(const v16bf*)(bbase + k0);
            acc = __builtin_amdgcn_wmma_f32_16x16x32_bf16(
                      false, a, false, bf, (short)0, acc, false, false);
        }
    }

    // ---- store: lane -> oc, VGPR r -> pixel (r + 8*half) -----------------
    {
        int oc = wv * 16 + row;
        float* op = out + (((size_t)(b * COUT + oc) * HW + y) * HW)
                        + x0 + half * 8;
        v4f lo = { acc[0], acc[1], acc[2], acc[3] };
        v4f hi = { acc[4], acc[5], acc[6], acc[7] };
        *(v4f*)op       = lo;
        *(v4f*)(op + 4) = hi;
    }
}

// ---------------------------------------------------------------------------
extern "C" void kernel_launch(void* const* d_in, const int* in_sizes, int n_in,
                              void* d_out, int out_size, void* d_ws, size_t ws_size,
                              hipStream_t stream)
{
    const float* x     = (const float*)d_in[0];
    const float* w_off = (const float*)d_in[1];
    const float* b_off = (const float*)d_in[2];
    const float* w     = (const float*)d_in[3];
    float* out = (float*)d_out;

    // workspace layout: [0, 147456) bf16 main weights, [147456, 184320) bf16 offset weights
    __bf16* wB    = (__bf16*)d_ws;
    __bf16* wOffB = (__bf16*)((char*)d_ws + (size_t)COUT * KDIM * sizeof(__bf16));

    pack_w_kernel   <<<288, 256, 0, stream>>>(w, wB);          // 128*576
    pack_woff_kernel<<< 72, 256, 0, stream>>>(w_off, wOffB);   // 32*576

    // 8 batches * 112 rows * 7 x-tiles of 16 pixels
    deform_wmma_kernel<<<6272, 256, 0, stream>>>(x, b_off, wOffB, wB, out);
}
